// YOLOLoss_24773371364043
// MI455X (gfx1250) — compile-verified, matrix operand order checked by reading, not run on hardware
//
#include <hip/hip_runtime.h>
#include <hip/hip_bf16.h>

// ---------------- CDNA5 types ----------------
typedef __attribute__((ext_vector_type(16))) _Float16 v16h;
typedef __attribute__((ext_vector_type(8)))  float    v8f;

// ---------------- problem constants ----------------
#define BATCH   16
#define NTGT    32          // targets per batch
#define NA      3           // anchors
#define NC      80          // classes
#define SS      80          // grid size
#define PLANE   (SS*SS)     // 6400
#define NCELLS  (BATCH*NA*PLANE)   // 307200
#define NCH     (5+NC)      // 85
#define STRIDEF 8.0f
#define IMGF    640.0f
#define EPSF    1e-7f
#define PI2     9.869604401089358f

// ws layout (4-byte units):
//  wsI[1] = numList (== cntObj: one append per obj cell)
//  wsF[2] = sum_obj softplus(conf)
//  wsF[3] = sum_all softplus(conf)
//  wsF[4] = sum CIoU
//  wsF[5] = sum class BCE
//  wsF[6] = sum_obj conf
//  +64 bytes: tidx int[NCELLS]
//  +64 + NCELLS*4: cellList int[1024]

__device__ __forceinline__ float softplusf_(float x) {
    return fmaxf(x, 0.0f) + log1pf(expf(-fabsf(x)));
}
__device__ __forceinline__ float sigmoidf_(float x) {
    return 1.0f / (1.0f + expf(-x));
}

// ---------------- kernel A: init workspace ----------------
__global__ void yolo_init(int* __restrict__ tidx, int* __restrict__ wsI) {
    int g = blockIdx.x * blockDim.x + threadIdx.x;
    if (g < NCELLS) tidx[g] = -1;
    if (g < 8) wsI[g] = 0;   // zero bit-pattern == 0.0f for the float slots too
}

// ---------------- kernel B: scatter targets -> cells ----------------
__global__ void yolo_scatter(const float* __restrict__ tbox,
                             const float* __restrict__ anchors,
                             int* __restrict__ tidx) {
    int t = blockIdx.x * blockDim.x + threadIdx.x;   // t = b*NTGT + n
    if (t >= BATCH * NTGT) return;
    int b = t >> 5;                                  // / NTGT (32)
    const float* tb = tbox + (size_t)t * 4;
    float x = tb[0], y = tb[1], w = tb[2], h = tb[3];
    int gi = min(max((int)floorf(x * (float)SS), 0), SS - 1);
    int gj = min(max((int)floorf(y * (float)SS), 0), SS - 1);
    float twp = w * IMGF, thp = h * IMGF;
    int best = 0; float bestIou = -1.0f;
#pragma unroll
    for (int a = 0; a < NA; ++a) {
        float aw = anchors[2 * a], ah = anchors[2 * a + 1];
        float inter = fminf(twp, aw) * fminf(thp, ah);
        float uni   = twp * thp + aw * ah - inter;
        float iou   = inter / uni;
        if (iou > bestIou) { bestIou = iou; best = a; }
    }
    int cell = ((b * NA + best) * SS + gj) * SS + gi;
    tidx[cell] = t;   // last-write-wins race == JAX scatter .set with dup indices
}

// ---------------- kernel C: dense conf pass, WMMA full-tile reduction ------
// Exactly 600 waves (75 blocks x 256 threads); each wave owns 512 cells
// (600*512 == NCELLS), so there are no bounds checks and EXEC is all 1s at
// WMMA issue. One v_wmma_f32_16x16x32_f16 sums softplus(conf) over all 512
// cells of the chunk: D = ones(16x32) x B(32x16) gives column sums replicated
// across rows (layout-agnostic for a total sum). Rare obj-cell terms are
// accumulated by direct atomics inside the obj branch (<=512 cells globally).
__global__ void yolo_dense_conf(const float* __restrict__ preds,
                                const int*   __restrict__ tidx,
                                int*   __restrict__ wsI,
                                float* __restrict__ wsF,
                                int*   __restrict__ cellList) {
    int lane  = threadIdx.x & 31;
    int wave  = (blockIdx.x * blockDim.x + threadIdx.x) >> 5;
    int chunk = wave * 512;

    v16h bsp, ones;
#pragma unroll
    for (int v = 0; v < 16; ++v) {
        int c      = chunk + v * 32 + lane;      // coalesced: 32 lanes contiguous
        int plane  = c / PLANE;
        int within = c - plane * PLANE;
        int b      = plane / NA;
        int a      = plane - b * NA;
        int off    = (b * (NA * NCH) + a * NCH + 4) * PLANE + within;  // conf chan
        float conf = preds[off];
        float sp   = softplusf_(conf);
        bsp[v]  = (_Float16)sp;
        ones[v] = (_Float16)1.0f;
        if (tidx[c] >= 0) {                      // obj cell (rare)
            int p = atomicAdd(&wsI[1], 1);       // numList == cntObj
            cellList[p] = c;
            atomicAdd(&wsF[2], sp);              // sum_obj softplus
            atomicAdd(&wsF[6], conf);            // sum_obj conf
        }
    }

    v8f acc = {};
    acc = __builtin_amdgcn_wmma_f32_16x16x32_f16(false, ones, false, bsp,
                                                 (short)0, acc, false, false);

    // C/D VGPR0: lanes 0-15 hold row M=0 = the 16 column sums.
    float s = acc[0];
    for (int off = 8; off >= 1; off >>= 1)
        s += __shfl_xor(s, off, 16);             // reduce within lanes 0..15
    if (lane == 0)
        atomicAdd(&wsF[3], s);                   // sum_all softplus
}

// ---------------- kernel D: obj cells (CIoU + class BCE) ----------------
__global__ void yolo_obj_cells(const float* __restrict__ preds,
                               const float* __restrict__ anchors,
                               const float* __restrict__ tbox,
                               const int*   __restrict__ tclsArr,
                               const int*   __restrict__ tidx,
                               const int*   __restrict__ cellList,
                               const int*   __restrict__ wsI,
                               float* __restrict__ wsF) {
    int idx = blockIdx.x * blockDim.x + threadIdx.x;
    int nl  = wsI[1];
    if (idx >= nl) return;

    int c      = cellList[idx];
    int t      = tidx[c];
    int plane  = c / PLANE;
    int within = c - plane * PLANE;
    int b      = plane / NA;
    int a      = plane - b * NA;
    int j      = within / SS;
    int i      = within - j * SS;
    int base   = (b * (NA * NCH) + a * NCH) * PLANE + within;

    // predicted box
    float lx = preds[base + 0 * PLANE];
    float ly = preds[base + 1 * PLANE];
    float lw = preds[base + 2 * PLANE];
    float lh = preds[base + 3 * PLANE];
    float aw = anchors[2 * a], ah = anchors[2 * a + 1];
    float pcx = (sigmoidf_(lx) + (float)i) * STRIDEF;
    float pcy = (sigmoidf_(ly) + (float)j) * STRIDEF;
    float pw  = aw * expf(lw);
    float ph  = ah * expf(lh);
    float px1 = pcx - pw * 0.5f, px2 = pcx + pw * 0.5f;
    float py1 = pcy - ph * 0.5f, py2 = pcy + ph * 0.5f;

    // target box
    const float* tb = tbox + (size_t)t * 4;
    float cx  = tb[0] * IMGF, cy = tb[1] * IMGF;
    float twp = tb[2] * IMGF, thp = tb[3] * IMGF;
    float tx1 = cx - twp * 0.5f, tx2 = cx + twp * 0.5f;
    float ty1 = cy - thp * 0.5f, ty2 = cy + thp * 0.5f;

    // CIoU
    float iw    = fmaxf(fminf(px2, tx2) - fmaxf(px1, tx1), 0.0f);
    float ih    = fmaxf(fminf(py2, ty2) - fmaxf(py1, ty1), 0.0f);
    float inter = iw * ih;
    float pa    = fmaxf(px2 - px1, 0.0f) * fmaxf(py2 - py1, 0.0f);
    float ta    = fmaxf(tx2 - tx1, 0.0f) * fmaxf(ty2 - ty1, 0.0f);
    float uni   = pa + ta - inter + EPSF;
    float iou   = inter / uni;
    float dx    = (px1 + px2) * 0.5f - (tx1 + tx2) * 0.5f;
    float dy    = (py1 + py2) * 0.5f - (ty1 + ty2) * 0.5f;
    float cd    = dx * dx + dy * dy;
    float ew    = fmaxf(px2, tx2) - fminf(px1, tx1);
    float eh    = fmaxf(py2, ty2) - fminf(py1, ty1);
    float ed    = ew * ew + eh * eh + EPSF;
    float pwc   = fmaxf(px2 - px1, EPSF), phc = fmaxf(py2 - py1, EPSF);
    float twc   = fmaxf(tx2 - tx1, EPSF), thc = fmaxf(ty2 - ty1, EPSF);
    float dv    = atanf(twc / thc) - atanf(pwc / phc);
    float vv    = (4.0f / PI2) * dv * dv;
    float alpha = vv / (1.0f - iou + vv + EPSF);
    float ciou  = iou - cd / ed - alpha * vv;
    atomicAdd(&wsF[4], ciou);

    // class BCE over 80 classes (target one-hot at tcls)
    int cls = tclsArr[t];
    float clsSum = 0.0f;
    for (int cc = 0; cc < NC; ++cc) {
        float l   = preds[base + (5 + cc) * PLANE];
        float tgt = (cc == cls) ? 1.0f : 0.0f;
        clsSum += softplusf_(l) - l * tgt;
    }
    atomicAdd(&wsF[5], clsSum);
}

// ---------------- kernel E: finalize ----------------
__global__ void yolo_finalize(const int* __restrict__ wsI,
                              const float* __restrict__ wsF,
                              float* __restrict__ out) {
    if (threadIdx.x == 0 && blockIdx.x == 0) {
        float cntObj   = (float)wsI[1];
        float cntNoobj = (float)NCELLS - cntObj;
        float mObj   = fmaxf(cntObj, 1.0f);
        float mNoobj = fmaxf(cntNoobj, 1.0f);
        float mCls   = fmaxf(cntObj * (float)NC, 1.0f);
        float sumObjBce   = wsF[2] - wsF[6];          // sum_obj (sp - conf)
        float sumNoobjBce = wsF[3] - wsF[2];          // sum_all sp - sum_obj sp
        float lossCiou      = 1.0f - wsF[4] / mObj;
        float lossConfObj   = sumObjBce / mObj;
        float lossConfNoobj = sumNoobjBce / mNoobj;
        float lossCls       = wsF[5] / mCls;
        out[0] = lossCiou + lossConfObj + 0.5f * lossConfNoobj + lossCls;
    }
}

extern "C" void kernel_launch(void* const* d_in, const int* in_sizes, int n_in,
                              void* d_out, int out_size, void* d_ws, size_t ws_size,
                              hipStream_t stream) {
    const float* preds   = (const float*)d_in[0];
    const float* anchors = (const float*)d_in[1];
    const int*   tcls    = (const int*)d_in[2];
    const float* tbox    = (const float*)d_in[3];
    float* out = (float*)d_out;

    int*   wsI      = (int*)d_ws;
    float* wsF      = (float*)d_ws;
    int*   tidx     = (int*)((char*)d_ws + 64);
    int*   cellList = (int*)((char*)d_ws + 64 + (size_t)NCELLS * 4);

    yolo_init<<<(NCELLS + 255) / 256, 256, 0, stream>>>(tidx, wsI);
    yolo_scatter<<<2, 256, 0, stream>>>(tbox, anchors, tidx);
    // exactly 600 waves: 75 blocks * 256 threads, 512 cells per wave
    yolo_dense_conf<<<75, 256, 0, stream>>>(preds, tidx, wsI, wsF, cellList);
    yolo_obj_cells<<<2, 256, 0, stream>>>(preds, anchors, tbox, tcls,
                                          tidx, cellList, wsI, wsF);
    yolo_finalize<<<1, 32, 0, stream>>>(wsI, wsF, out);
}
